// FrameTransformerDecoder_16930761080904
// MI455X (gfx1250) — matrix operand order repaired and checked by reading, not compiled, not run on hardware
//
#include <hip/hip_runtime.h>
#include <hip/hip_bf16.h>

// ---------------------------------------------------------------------------
// Problem constants (from reference): B=1, C=8, OC=8, F=1024, W=512, H=8, d=128
// ---------------------------------------------------------------------------
#define FDIM   1024
#define WDIM   512
#define PLANE  (FDIM * WDIM)          // 524288 elements per channel plane
#define NMAT   64                     // OC * H independent attention matrices
#define SEQ    512                    // attention sequence length (W)
#define DHEAD  128                    // head dim
#define QKELEM (SEQ * SEQ)            // 262144 per matrix
#define QVELEM (DHEAD * SEQ)          // 65536 per matrix (q/k/v/a slice)

typedef __attribute__((ext_vector_type(16))) __bf16 v16bf;
typedef __attribute__((ext_vector_type(8)))  float  v8f;
typedef __attribute__((ext_vector_type(4)))  float  f32x4;

// ---------------------------------------------------------------------------
// LayerNorm over F axis, per (channel, w). Supports up to 3 concatenated
// channel sources (for h = [x | z1 | z2]) without materializing the concat.
// grid: (WDIM/64, Ctot), block: 256 (64 w-lanes x 4 f-groups)
// ---------------------------------------------------------------------------
__global__ __launch_bounds__(256) void ln_kernel(
    const float* __restrict__ s0, const float* __restrict__ s1,
    const float* __restrict__ s2, int c0, int c1,
    const float* __restrict__ g, const float* __restrict__ b,
    float* __restrict__ out)
{
    int c  = blockIdx.y;
    int w0 = blockIdx.x * 64;
    int wl = threadIdx.x & 63;
    int fl = threadIdx.x >> 6;     // 0..3

    const float* src;
    int cl = c;
    if (cl < c0)            { src = s0; }
    else { cl -= c0;
        if (cl < c1)        { src = s1; }
        else { cl -= c1;      src = s2; } }

    const float* base = src + (size_t)cl * PLANE + w0 + wl;

    float sum = 0.f, sq = 0.f;
    for (int f = fl; f < FDIM; f += 4) {
        float v = base[(size_t)f * WDIM];
        sum += v; sq += v * v;
    }
    __shared__ float lsum[4][64];
    __shared__ float lsq[4][64];
    lsum[fl][wl] = sum; lsq[fl][wl] = sq;
    __syncthreads();
    float ts = lsum[0][wl] + lsum[1][wl] + lsum[2][wl] + lsum[3][wl];
    float tq = lsq[0][wl]  + lsq[1][wl]  + lsq[2][wl]  + lsq[3][wl];
    float mu   = ts * (1.0f / FDIM);
    float var  = tq * (1.0f / FDIM) - mu * mu;
    float rstd = rsqrtf(var + 1e-8f);

    float* ob = out + (size_t)c * PLANE + w0 + wl;
    const float* gc = g + (size_t)c * FDIM;
    const float* bc = b + (size_t)c * FDIM;
    for (int f = fl; f < FDIM; f += 4) {
        float v = base[(size_t)f * WDIM];
        ob[(size_t)f * WDIM] = (v - mu) * rstd * gc[f] + bc[f];
    }
}

// ---------------------------------------------------------------------------
// 1x3 conv along W (pad 1). Weights (Cout, Cin, 1, 3); o = blockIdx.y so the
// weight reads are wave-uniform (scalar loads).
// mode 0: plane layout  out[o][f][w]            (d-major, for V / A / z)
// mode 1: head layout   out[o][h][w][dd]        (seq-major, for Q / K so the
//         QK WMMA fragment loads are 128-bit contiguous in d)
// grid: (PLANE/256, Cout), block 256
// ---------------------------------------------------------------------------
__global__ __launch_bounds__(256) void conv1x3_kernel(
    const float* __restrict__ in, const float* __restrict__ wgt,
    float* __restrict__ out, int Cin, int mode)
{
    int o   = blockIdx.y;
    int idx = blockIdx.x * 256 + threadIdx.x;       // < PLANE
    int f   = idx >> 9;
    int w   = idx & 511;

    const float* wr = wgt + (size_t)o * Cin * 3;
    float acc = 0.f;
    for (int ci = 0; ci < Cin; ++ci) {
        const float* row = in + (size_t)ci * PLANE + (size_t)f * WDIM;
        float l = (w > 0)        ? row[w - 1] : 0.f;
        float m = row[w];
        float r = (w < WDIM - 1) ? row[w + 1] : 0.f;
        acc = fmaf(wr[3 * ci + 0], l, acc);
        acc = fmaf(wr[3 * ci + 1], m, acc);
        acc = fmaf(wr[3 * ci + 2], r, acc);
    }
    if (mode == 0) {
        out[(size_t)o * PLANE + idx] = acc;
    } else {
        int h  = f >> 7;            // head
        int dd = f & 127;           // d within head
        out[(size_t)o * PLANE + (size_t)h * QVELEM + (size_t)w * DHEAD + dd] = acc;
    }
}

// ---------------------------------------------------------------------------
// QK^T with bf16 WMMA, fused scale (1/sqrt(d)) and up-to-3 fp32 bias adds.
// q/k are in head layout [mat][i][d] -> all fragment loads are b128.
// One wave computes one 16x16 tile of the 512x512 output; K-loop = 4 WMMAs.
// grid: NMAT*32*32/8 = 8192 blocks of 256 threads (8 waves).
// ---------------------------------------------------------------------------
__global__ __launch_bounds__(256) void qk_wmma_kernel(
    const float* __restrict__ q, const float* __restrict__ k,
    const float* __restrict__ b0, const float* __restrict__ b1,
    const float* __restrict__ b2, float* __restrict__ out)
{
    int lane = threadIdx.x & 31;
    int gw   = blockIdx.x * 8 + (threadIdx.x >> 5);
    int mat  = gw >> 10;                 // 0..63
    int t    = gw & 1023;
    int i0   = (t >> 5) << 4;
    int j0   = (t & 31) << 4;

    const float* qb = q + (size_t)mat * QVELEM;
    const float* kb = k + (size_t)mat * QVELEM;

    int m  = lane & 15;
    int hi = lane >> 4;                  // 0 or 1

    v8f acc = {0.f, 0.f, 0.f, 0.f, 0.f, 0.f, 0.f, 0.f};
    for (int k0 = 0; k0 < DHEAD; k0 += 32) {
        // A (16x32, row = lane%16): lanes<16 hold K 0..15, lanes>=16 hold K 16..31
        const float* arow = qb + (size_t)(i0 + m) * DHEAD + k0 + (hi << 4);
        f32x4 a0 = *(const f32x4*)(arow + 0);
        f32x4 a1 = *(const f32x4*)(arow + 4);
        f32x4 a2 = *(const f32x4*)(arow + 8);
        f32x4 a3 = *(const f32x4*)(arow + 12);
        // B (32x16, col = lane%16): lanes<16 hold K 0..15, lanes>=16 hold K 16..31
        const float* brow = kb + (size_t)(j0 + m) * DHEAD + k0 + (hi << 4);
        f32x4 c0 = *(const f32x4*)(brow + 0);
        f32x4 c1 = *(const f32x4*)(brow + 4);
        f32x4 c2 = *(const f32x4*)(brow + 8);
        f32x4 c3 = *(const f32x4*)(brow + 12);

        v16bf A, B;
#pragma unroll
        for (int e = 0; e < 4; ++e) {
            A[e]      = (__bf16)a0[e];
            A[4 + e]  = (__bf16)a1[e];
            A[8 + e]  = (__bf16)a2[e];
            A[12 + e] = (__bf16)a3[e];
            B[e]      = (__bf16)c0[e];
            B[4 + e]  = (__bf16)c1[e];
            B[8 + e]  = (__bf16)c2[e];
            B[12 + e] = (__bf16)c3[e];
        }
        acc = __builtin_amdgcn_wmma_f32_16x16x32_bf16(
            false, A, false, B, (short)0, acc, false, false);
    }

    const float scale = 0.08838834764831845f;    // 1/sqrt(128)
    size_t mbase = (size_t)mat * QKELEM;
#pragma unroll
    for (int r = 0; r < 8; ++r) {
        int i = i0 + r + (hi << 3);              // C: VGPR r -> M=r (+8 for high half)
        size_t idx = mbase + (size_t)i * SEQ + j0 + m;
        float v = acc[r] * scale + b0[idx] + b1[idx];
        if (b2) v += b2[idx];
        out[idx] = v;
    }
}

// ---------------------------------------------------------------------------
// softmax(QK) @ V with bf16 WMMA.  One block per (matrix, 16-row tile):
// stage 16x512 f32 scores in LDS (b128 copies), softmax in LDS, then 8 waves
// each produce a 16x16 slice of the 16x128 output (16 WMMAs per wave).
// V is d-major so B-fragment loads are b128; A comes from LDS as ds_load_b128.
// Output written transposed as [d][i] planes (layout the wo conv consumes).
// grid: NMAT*32 = 2048 blocks of 256 threads.
// ---------------------------------------------------------------------------
__global__ __launch_bounds__(256) void av_wmma_kernel(
    const float* __restrict__ qk, const float* __restrict__ v,
    float* __restrict__ a)
{
    __shared__ float S[16][SEQ];      // 32 KB
    __shared__ float red[16][16];

    int tid  = threadIdx.x;
    int lane = tid & 31;
    int mat  = blockIdx.x >> 5;
    int i0   = (blockIdx.x & 31) << 4;

    const f32x4* sb4 = (const f32x4*)(qk + (size_t)mat * QKELEM + (size_t)i0 * SEQ);
    f32x4* S4 = (f32x4*)S;
    for (int idx = tid; idx < 16 * SEQ / 4; idx += 256)
        S4[idx] = sb4[idx];
    __syncthreads();

    // --- softmax over each of the 16 rows (16 threads per row) ---
    int r    = tid >> 4;
    int tcol = tid & 15;
    float mx = -3.402823466e38f;
    for (int j = tcol; j < SEQ; j += 16) mx = fmaxf(mx, S[r][j]);
    red[r][tcol] = mx;
    __syncthreads();
    float rm = red[r][0];
#pragma unroll
    for (int kk = 1; kk < 16; ++kk) rm = fmaxf(rm, red[r][kk]);
    __syncthreads();
    float sum = 0.f;
    for (int j = tcol; j < SEQ; j += 16) {
        float e = __expf(S[r][j] - rm);
        S[r][j] = e; sum += e;
    }
    red[r][tcol] = sum;
    __syncthreads();
    float rs = 0.f;
#pragma unroll
    for (int kk = 0; kk < 16; ++kk) rs += red[r][kk];
    float rc = 1.0f / rs;
    for (int j = tcol; j < SEQ; j += 16) S[r][j] *= rc;
    __syncthreads();

    // --- P(16x512) @ V(512x128), wave wv owns d-columns [wv*16, wv*16+16) ---
    const float* vb = v + (size_t)mat * QVELEM;
    int n0 = (tid >> 5) << 4;
    int m  = lane & 15;
    int hi = lane >> 4;

    v8f acc = {0.f, 0.f, 0.f, 0.f, 0.f, 0.f, 0.f, 0.f};
    for (int k0 = 0; k0 < SEQ; k0 += 32) {
        // A from LDS: rows contiguous in k -> four b128
        const float* sr = &S[m][k0 + (hi << 4)];
        f32x4 a0 = *(const f32x4*)(sr + 0);
        f32x4 a1 = *(const f32x4*)(sr + 4);
        f32x4 a2 = *(const f32x4*)(sr + 8);
        f32x4 a3 = *(const f32x4*)(sr + 12);
        // B from V (d-major): contiguous in k -> four b128
        const float* vr = vb + (size_t)(n0 + m) * SEQ + k0 + (hi << 4);
        f32x4 c0 = *(const f32x4*)(vr + 0);
        f32x4 c1 = *(const f32x4*)(vr + 4);
        f32x4 c2 = *(const f32x4*)(vr + 8);
        f32x4 c3 = *(const f32x4*)(vr + 12);

        v16bf A, B;
#pragma unroll
        for (int e = 0; e < 4; ++e) {
            A[e]      = (__bf16)a0[e];
            A[4 + e]  = (__bf16)a1[e];
            A[8 + e]  = (__bf16)a2[e];
            A[12 + e] = (__bf16)a3[e];
            B[e]      = (__bf16)c0[e];
            B[4 + e]  = (__bf16)c1[e];
            B[8 + e]  = (__bf16)c2[e];
            B[12 + e] = (__bf16)c3[e];
        }
        acc = __builtin_amdgcn_wmma_f32_16x16x32_bf16(
            false, A, false, B, (short)0, acc, false, false);
    }

    float* ob = a + (size_t)mat * QVELEM;
#pragma unroll
    for (int rr = 0; rr < 8; ++rr) {
        int i = i0 + rr + (hi << 3);
        ob[(size_t)(n0 + m) * SEQ + i] = acc[rr];   // transposed [d][i]
    }
}

// ---------------------------------------------------------------------------
// 3x3 conv (pad 1,1) + fused Squared-ReLU.  grid: (PLANE/256, Cout)
// ---------------------------------------------------------------------------
__global__ __launch_bounds__(256) void conv3x3_sqrelu_kernel(
    const float* __restrict__ in, const float* __restrict__ wgt,
    float* __restrict__ out, int Cin)
{
    int o   = blockIdx.y;
    int idx = blockIdx.x * 256 + threadIdx.x;
    int f   = idx >> 9;
    int w   = idx & 511;

    const float* wr = wgt + (size_t)o * Cin * 9;
    float acc = 0.f;
    for (int ci = 0; ci < Cin; ++ci) {
        const float* w9   = wr + ci * 9;
        const float* rowc = in + (size_t)ci * PLANE + (size_t)f * WDIM;
#pragma unroll
        for (int df = 0; df < 3; ++df) {
            int ff = f + df - 1;
            if (ff < 0 || ff >= FDIM) continue;
            const float* row = rowc + (df - 1) * WDIM;
            float l = (w > 0)        ? row[w - 1] : 0.f;
            float c = row[w];
            float rr = (w < WDIM - 1) ? row[w + 1] : 0.f;
            acc = fmaf(w9[df * 3 + 0], l,  acc);
            acc = fmaf(w9[df * 3 + 1], c,  acc);
            acc = fmaf(w9[df * 3 + 2], rr, acc);
        }
    }
    float rv = fmaxf(acc, 0.f);
    out[(size_t)o * PLANE + idx] = rv * rv;
}

// ---------------------------------------------------------------------------
// 3x3 conv (pad 1,1) + fused residual add (out = resid + conv)
// ---------------------------------------------------------------------------
__global__ __launch_bounds__(256) void conv3x3_add_kernel(
    const float* __restrict__ in, const float* __restrict__ wgt,
    const float* __restrict__ resid, float* __restrict__ out, int Cin)
{
    int o   = blockIdx.y;
    int idx = blockIdx.x * 256 + threadIdx.x;
    int f   = idx >> 9;
    int w   = idx & 511;

    const float* wr = wgt + (size_t)o * Cin * 9;
    float acc = 0.f;
    for (int ci = 0; ci < Cin; ++ci) {
        const float* w9   = wr + ci * 9;
        const float* rowc = in + (size_t)ci * PLANE + (size_t)f * WDIM;
#pragma unroll
        for (int df = 0; df < 3; ++df) {
            int ff = f + df - 1;
            if (ff < 0 || ff >= FDIM) continue;
            const float* row = rowc + (df - 1) * WDIM;
            float l = (w > 0)        ? row[w - 1] : 0.f;
            float c = row[w];
            float rr = (w < WDIM - 1) ? row[w + 1] : 0.f;
            acc = fmaf(w9[df * 3 + 0], l,  acc);
            acc = fmaf(w9[df * 3 + 1], c,  acc);
            acc = fmaf(w9[df * 3 + 2], rr, acc);
        }
    }
    out[(size_t)o * PLANE + idx] = acc + resid[(size_t)o * PLANE + idx];
}

// ---------------------------------------------------------------------------
extern "C" void kernel_launch(void* const* d_in, const int* in_sizes, int n_in,
                              void* d_out, int out_size, void* d_ws, size_t ws_size,
                              hipStream_t stream) {
    (void)in_sizes; (void)n_in; (void)out_size; (void)ws_size;

    const float* x     = (const float*)d_in[0];
    const float* skip  = (const float*)d_in[1];
    const float* pqk1  = (const float*)d_in[2];
    const float* pqk2  = (const float*)d_in[3];
    const float* sqk   = (const float*)d_in[4];
    const float* g1    = (const float*)d_in[5];
    const float* b1    = (const float*)d_in[6];
    const float* wq1   = (const float*)d_in[7];
    const float* wk1   = (const float*)d_in[8];
    const float* wv1   = (const float*)d_in[9];
    const float* wo1   = (const float*)d_in[10];
    const float* g2    = (const float*)d_in[11];
    const float* b2    = (const float*)d_in[12];
    const float* wq2   = (const float*)d_in[13];
    const float* wk2   = (const float*)d_in[14];
    const float* wv2   = (const float*)d_in[15];
    const float* wo2   = (const float*)d_in[16];
    const float* g3    = (const float*)d_in[17];
    const float* b3    = (const float*)d_in[18];
    const float* wc1   = (const float*)d_in[19];
    const float* wc2   = (const float*)d_in[20];

    float* out0 = (float*)d_out;                         // (8,1024,512)
    float* pq1  = out0 + (size_t)8 * PLANE;              // (64,512,512)
    float* pq2  = pq1 + (size_t)NMAT * QKELEM;           // (64,512,512)

    // Workspace layout (floats, P = PLANE):
    //  [0,24P)  ln output (nx:8ch, nh:16ch, n3:24ch — sequential reuse)
    //  [24,32P) q   [32,40P) k   [40,48P) v   [48,56P) a
    //  [56,64P) z1  [64,72P) z2
    //  t (32P) reuses [24,56P) after attention 2 is done.
    float* wsf = (float*)d_ws;
    float* lnb = wsf;
    float* qb  = wsf + (size_t)24 * PLANE;
    float* kb  = wsf + (size_t)32 * PLANE;
    float* vb  = wsf + (size_t)40 * PLANE;
    float* ab  = wsf + (size_t)48 * PLANE;
    float* z1  = wsf + (size_t)56 * PLANE;
    float* z2  = wsf + (size_t)64 * PLANE;
    float* tb  = wsf + (size_t)24 * PLANE;               // 32P, reuse

    dim3 blk(256);
    dim3 gConv(PLANE / 256, 8);       // Cout = 8 projections
    dim3 gQK(NMAT * 32 * 32 / 8);     // 8192 blocks (8 waves each -> 1 tile/wave)
    dim3 gAV(NMAT * 32);              // 2048 blocks

    // ---- block 1: self-attention on nx ----
    ln_kernel<<<dim3(WDIM / 64, 8), blk, 0, stream>>>(x, nullptr, nullptr, 8, 0, g1, b1, lnb);
    conv1x3_kernel<<<gConv, blk, 0, stream>>>(lnb, wq1, qb, 8, 1);
    conv1x3_kernel<<<gConv, blk, 0, stream>>>(lnb, wk1, kb, 8, 1);
    conv1x3_kernel<<<gConv, blk, 0, stream>>>(lnb, wv1, vb, 8, 0);
    qk_wmma_kernel<<<gQK, blk, 0, stream>>>(qb, kb, pqk1, sqk, nullptr, pq1);
    av_wmma_kernel<<<gAV, blk, 0, stream>>>(pq1, vb, ab);
    conv1x3_kernel<<<gConv, blk, 0, stream>>>(ab, wo1, z1, 8, 0);

    // ---- block 2: cross-attention, query = LN([x,z1]), memory = skip ----
    ln_kernel<<<dim3(WDIM / 64, 16), blk, 0, stream>>>(x, z1, nullptr, 8, 8, g2, b2, lnb);
    conv1x3_kernel<<<gConv, blk, 0, stream>>>(lnb, wq2, qb, 16, 1);
    conv1x3_kernel<<<gConv, blk, 0, stream>>>(skip, wk2, kb, 8, 1);
    conv1x3_kernel<<<gConv, blk, 0, stream>>>(skip, wv2, vb, 8, 0);
    qk_wmma_kernel<<<gQK, blk, 0, stream>>>(qb, kb, sqk, pq1, pqk2, pq2);
    av_wmma_kernel<<<gAV, blk, 0, stream>>>(pq2, vb, ab);
    conv1x3_kernel<<<gConv, blk, 0, stream>>>(ab, wo2, z2, 8, 0);

    // ---- conv MLP: LN([x,z1,z2]) -> 3x3 conv -> sqrelu -> 3x3 conv (+x) ----
    ln_kernel<<<dim3(WDIM / 64, 24), blk, 0, stream>>>(x, z1, z2, 8, 8, g3, b3, lnb);
    conv3x3_sqrelu_kernel<<<dim3(PLANE / 256, 32), blk, 0, stream>>>(lnb, wc1, tb, 24);
    conv3x3_add_kernel<<<dim3(PLANE / 256, 8), blk, 0, stream>>>(tb, wc2, x, out0, 8 * 4);
}
